// Swinv2Attention_17789754540126
// MI455X (gfx1250) — matrix operand order repaired and checked by reading, not compiled, not run on hardware
//
#include <hip/hip_runtime.h>
#include <hip/hip_bf16.h>

// ---------- WMMA plumbing (CDNA5 gfx1250, wave32) ----------
typedef __attribute__((ext_vector_type(16))) _Float16 v16h;
typedef __attribute__((ext_vector_type(8)))  float    v8f;

union Frag {
    v16h v;
    uint4 q[2];
    _Float16 e[16];
};

__device__ __forceinline__ v8f wmma_f16(const Frag& a, const Frag& b, v8f c) {
    // D = A(16x32 f16) * B(32x16 f16) + C(16x16 f32)
    return __builtin_amdgcn_wmma_f32_16x16x32_f16(
        /*neg_a=*/false, a.v, /*neg_b=*/false, b.v,
        /*c_mod=*/(short)0, c, /*reuse_a=*/false, /*reuse_b=*/false);
}

// Geometry constants
#define MT    200704          // B*S = 4096*49 rows
#define DIMK  384
#define ASTR  40              // LDS row stride (halves) for 32-wide K tiles: bank-conflict free
#define PSTR  72              // LDS row stride (halves) for 64-wide tiles: bank-conflict free

// ---------------------------------------------------------------------------
// Generic 128x128x(K=384) GEMM block:  C[m][n] = sum_k A[m][k] * W[n][k] (+bias[n])
// A: row-major [*,384] (f32 or f16).  W: row-major f32 [*,384].  256 thr = 8 waves.
// Each wave computes a 64x32 sub-tile: 4 M-tiles x 2 N-tiles of 16x16.
// ---------------------------------------------------------------------------
template<bool AF16, bool OF32>
__device__ __forceinline__ void gemm_block(
    const void* __restrict__ Ab,      // A block base (row 0 of this M-block)
    const float* __restrict__ Wrow,   // W base, pre-offset to this N-block's first row
    const float* __restrict__ bias,   // per-local-n bias (128 entries) or nullptr
    void* __restrict__ OutBase,       // output element (0,0) of this block; ld = 384
    _Float16* As, _Float16* Bs)
{
    const int tid  = threadIdx.x;
    const int lane = tid & 31;
    const int wave = tid >> 5;
    const int wm   = (wave >> 2) * 64;   // wave M offset: 0 / 64
    const int wn   = (wave & 3) * 32;    // wave N offset: 0..96
    const int lhi  = lane >> 4;          // lane half (0/1)
    const int llo  = lane & 15;

    v8f zero = {};
    v8f acc[4][2];
    for (int i = 0; i < 4; ++i)
        for (int j = 0; j < 2; ++j) acc[i][j] = zero;

    for (int kb = 0; kb < DIMK; kb += 32) {
        // ---- stage A tile: 128 rows x 32 cols -> f16 LDS ----
        if constexpr (AF16) {
            const _Float16* A = (const _Float16*)Ab;
            for (int it = 0; it < 2; ++it) {
                int idx = tid + it * 256;             // 0..511
                int r = idx >> 2, c8 = (idx & 3) * 8;
                uint4 d = *(const uint4*)(A + (size_t)r * DIMK + kb + c8);
                *(uint4*)(As + r * ASTR + c8) = d;
            }
        } else {
            const float* A = (const float*)Ab;
            for (int it = 0; it < 4; ++it) {
                int idx = tid + it * 256;             // 0..1023
                int r = idx >> 3, c4 = (idx & 7) * 4;
                float4 d = *(const float4*)(A + (size_t)r * DIMK + kb + c4);
                _Float16* s = As + r * ASTR + c4;
                s[0] = (_Float16)d.x; s[1] = (_Float16)d.y;
                s[2] = (_Float16)d.z; s[3] = (_Float16)d.w;
            }
        }
        // ---- stage W tile: 128 rows x 32 cols (always f32 source) ----
        for (int it = 0; it < 4; ++it) {
            int idx = tid + it * 256;
            int r = idx >> 3, c4 = (idx & 7) * 4;
            float4 d = *(const float4*)(Wrow + (size_t)r * DIMK + kb + c4);
            _Float16* s = Bs + r * ASTR + c4;
            s[0] = (_Float16)d.x; s[1] = (_Float16)d.y;
            s[2] = (_Float16)d.z; s[3] = (_Float16)d.w;
        }
        __syncthreads();

        // ---- fragment loads ----
        Frag aF[4], bF[2];
        for (int mt = 0; mt < 4; ++mt) {
            // A layout: lane=M; lanes 0-15 hold K 0-7 & 16-23, lanes 16-31 hold K 8-15 & 24-31
            const _Float16* src = As + (wm + mt * 16 + llo) * ASTR + (lhi << 3);
            aF[mt].q[0] = *(const uint4*)src;
            aF[mt].q[1] = *(const uint4*)(src + 16);
        }
        for (int nt = 0; nt < 2; ++nt) {
            // B layout: lane=N; lanes 0-15 hold K 0-7 & 16-23, lanes 16-31 hold K 8-15 & 24-31
            const _Float16* src = Bs + (wn + nt * 16 + llo) * ASTR + (lhi << 3);
            bF[nt].q[0] = *(const uint4*)src;
            bF[nt].q[1] = *(const uint4*)(src + 16);
        }
        for (int mt = 0; mt < 4; ++mt)
            for (int nt = 0; nt < 2; ++nt)
                acc[mt][nt] = wmma_f16(aF[mt], bF[nt], acc[mt][nt]);
        __syncthreads();
    }

    // ---- epilogue: C layout lane=N, VGPR r -> M = r + 8*(lane>>4) ----
    for (int mt = 0; mt < 4; ++mt) {
        for (int nt = 0; nt < 2; ++nt) {
            int nl = wn + nt * 16 + llo;
            float bvv = bias ? bias[nl] : 0.f;
            for (int r = 0; r < 8; ++r) {
                int m = wm + mt * 16 + r + (lhi << 3);
                float val = acc[mt][nt][r] + bvv;
                if constexpr (OF32)
                    ((float*)OutBase)[(size_t)m * DIMK + nl] = val;
                else
                    ((_Float16*)OutBase)[(size_t)m * DIMK + nl] = (_Float16)val;
            }
        }
    }
}

// ---------------------------------------------------------------------------
// Kernel 1: fused QKV projection. grid = (1568, 9): y selects which 128-col
// slab of the virtual 1152-wide [Wq;Wk;Wv] output. 384/128 = 3 slabs each.
// ---------------------------------------------------------------------------
__global__ __launch_bounds__(256) void qkv_kernel(
    const float* __restrict__ X,
    const float* __restrict__ Wq, const float* __restrict__ Wk,
    const float* __restrict__ Wv,
    const float* __restrict__ bq, const float* __restrict__ bv,
    _Float16* __restrict__ qb, _Float16* __restrict__ kb, _Float16* __restrict__ vb)
{
    __shared__ _Float16 As[128 * ASTR];
    __shared__ _Float16 Bs[128 * ASTR];
    size_t mBase = (size_t)blockIdx.x * 128;
    int ty    = blockIdx.y;
    int which = ty / 3;
    int nlb   = (ty % 3) * 128;
    const float* W    = (which == 0) ? Wq : (which == 1) ? Wk : Wv;
    const float* bias = (which == 0) ? (bq + nlb) : (which == 2) ? (bv + nlb) : nullptr;
    _Float16* out = ((which == 0) ? qb : (which == 1) ? kb : vb) + mBase * DIMK + nlb;
    gemm_block<false, false>(X + mBase * DIMK, W + (size_t)nlb * DIMK, bias, out, As, Bs);
}

// ---------------------------------------------------------------------------
// Kernel 3: output projection  out = ctx @ Wp^T + bp  (f16 in, f32 out)
// ---------------------------------------------------------------------------
__global__ __launch_bounds__(256) void proj_kernel(
    const _Float16* __restrict__ ctx, const float* __restrict__ Wp,
    const float* __restrict__ bp, float* __restrict__ out)
{
    __shared__ _Float16 As[128 * ASTR];
    __shared__ _Float16 Bs[128 * ASTR];
    size_t mBase = (size_t)blockIdx.x * 128;
    int nlb = blockIdx.y * 128;
    gemm_block<true, true>(ctx + mBase * DIMK, Wp + (size_t)nlb * DIMK,
                           bp + nlb, out + mBase * DIMK + nlb, As, Bs);
}

// ---------------------------------------------------------------------------
// Kernel 0: CPB MLP -> padded relative-position bias [12][64][64] f32
// bias = 16*sigmoid( relu(T @ w0^T + b0) @ w1^T ) gathered by rel_pos_index.
// ---------------------------------------------------------------------------
__global__ __launch_bounds__(256) void cpb_kernel(
    const float* __restrict__ table, const float* __restrict__ w0,
    const float* __restrict__ b0, const float* __restrict__ w1,
    const int* __restrict__ idx, float* __restrict__ biasb)
{
    __shared__ float tbl[169 * 12];
    int tid = threadIdx.x;
    for (int o = tid; o < 169 * 12; o += 256) {
        int t = o / 12, hh = o % 12;
        float x0 = table[t * 2], x1 = table[t * 2 + 1];
        float acc = 0.f;
        for (int c = 0; c < 512; ++c) {
            float pre = fmaf(x0, w0[c * 2], fmaf(x1, w0[c * 2 + 1], b0[c]));
            acc = fmaf(fmaxf(pre, 0.f), w1[hh * 512 + c], acc);
        }
        tbl[o] = 16.f / (1.f + __expf(-acc));
    }
    __syncthreads();
    for (int o = tid; o < 12 * 64 * 64; o += 256) {
        int hh = o >> 12, m = (o >> 6) & 63, n = o & 63;
        float v = 0.f;
        if (m < 49 && n < 49) v = tbl[idx[m * 49 + n] * 12 + hh];
        biasb[o] = v;
    }
}

// ---------------------------------------------------------------------------
// Kernel 2: cosine attention per (window b, head h). 128 thr = 4 waves, each
// wave owns a 16-row query block of the 64-padded window.
// ---------------------------------------------------------------------------
__global__ __launch_bounds__(128) void attn_kernel(
    const _Float16* __restrict__ qb, const _Float16* __restrict__ kb,
    const _Float16* __restrict__ vb, const float* __restrict__ biasb,
    const float* __restrict__ logit_scale, _Float16* __restrict__ ctx)
{
    __shared__ float qn[64], kn[64];
    __shared__ _Float16 Vt[32 * PSTR];   // V transposed: [hd][tok]
    __shared__ _Float16 Pb[64 * PSTR];   // softmax probabilities (f16)

    const int b = blockIdx.x, h = blockIdx.y;
    const int tid = threadIdx.x, lane = tid & 31, w = tid >> 5;
    const int lhi = lane >> 4, llo = lane & 15;
    const size_t rowbase = (size_t)b * 49 * DIMK + h * 32; // token-0 element offset

    // ---- reciprocal L2 norms of q and k rows ----
    {
        int i = (tid < 64) ? tid : tid - 64;
        const _Float16* src = (tid < 64) ? qb : kb;
        float* dst = (tid < 64) ? qn : kn;
        float inv = 0.f;
        if (i < 49) {
            const _Float16* p = src + rowbase + (size_t)i * DIMK;
            float ss = 0.f;
            for (int c = 0; c < 32; ++c) { float x = (float)p[c]; ss = fmaf(x, x, ss); }
            inv = 1.f / fmaxf(sqrtf(ss), 1e-6f);
        }
        dst[i] = inv;
    }
    // ---- stage V transposed into LDS (zero-padded to 64 tokens) ----
    for (int i2 = tid; i2 < 64 * 32; i2 += 128) {
        int tok = i2 >> 5, hd = i2 & 31;
        _Float16 val = (_Float16)0.f;
        if (tok < 49) val = vb[rowbase + (size_t)tok * DIMK + hd];
        Vt[hd * PSTR + tok] = val;
    }
    __syncthreads();

    // ---- Q A-fragment for this wave's 16 query rows (K = HD = 32) ----
    Frag aq;
    {
        int s = w * 16 + llo;
        if (s < 49) {
            const _Float16* p = qb + rowbase + (size_t)s * DIMK + (lhi << 3);
            aq.q[0] = *(const uint4*)p;
            aq.q[1] = *(const uint4*)(p + 16);
        } else {
            aq.q[0] = make_uint4(0, 0, 0, 0);
            aq.q[1] = make_uint4(0, 0, 0, 0);
        }
    }
    // ---- logits: 4 tiles of 16x16, one WMMA each (B = K^T from k rows) ----
    v8f zero = {};
    v8f acc[4];
    for (int t = 0; t < 4; ++t) {
        Frag bk;
        int s = t * 16 + llo;
        if (s < 49) {
            const _Float16* p = kb + rowbase + (size_t)s * DIMK + (lhi << 3);
            bk.q[0] = *(const uint4*)p;
            bk.q[1] = *(const uint4*)(p + 16);
        } else {
            bk.q[0] = make_uint4(0, 0, 0, 0);
            bk.q[1] = make_uint4(0, 0, 0, 0);
        }
        acc[t] = wmma_f16(aq, bk, zero);
    }

    // ---- scale + norms + bias + column mask ----
    const float scale = __expf(logit_scale[h]);
    const float* bh = biasb + h * 64 * 64;
    for (int t = 0; t < 4; ++t) {
        int n = t * 16 + llo;
        float knv = kn[n];
        for (int r = 0; r < 8; ++r) {
            int m = w * 16 + r + (lhi << 3);
            float l = acc[t][r] * qn[m] * knv * scale + bh[m * 64 + n];
            if (n >= 49) l = -3.0e38f;
            acc[t][r] = l;
        }
    }
    // ---- row softmax: row M lives across 16 lanes (same lane-half), 4 tiles ----
    for (int r = 0; r < 8; ++r) {
        float mx = fmaxf(fmaxf(acc[0][r], acc[1][r]), fmaxf(acc[2][r], acc[3][r]));
        for (int d = 1; d < 16; d <<= 1) mx = fmaxf(mx, __shfl_xor(mx, d, 32));
        float e0 = __expf(acc[0][r] - mx);
        float e1 = __expf(acc[1][r] - mx);
        float e2 = __expf(acc[2][r] - mx);
        float e3 = __expf(acc[3][r] - mx);
        float sum = e0 + e1 + e2 + e3;
        for (int d = 1; d < 16; d <<= 1) sum += __shfl_xor(sum, d, 32);
        float rinv = 1.f / sum;
        int m = w * 16 + r + (lhi << 3);
        Pb[m * PSTR + 0 * 16 + llo] = (_Float16)(e0 * rinv);
        Pb[m * PSTR + 1 * 16 + llo] = (_Float16)(e1 * rinv);
        Pb[m * PSTR + 2 * 16 + llo] = (_Float16)(e2 * rinv);
        Pb[m * PSTR + 3 * 16 + llo] = (_Float16)(e3 * rinv);
    }
    __syncthreads();

    // ---- out = P(16x64) @ V(64x32): 2 K-tiles x 2 N-tiles WMMAs ----
    v8f o[2]; o[0] = zero; o[1] = zero;
    for (int kt = 0; kt < 2; ++kt) {
        Frag ap;
        const _Float16* pp = Pb + (w * 16 + llo) * PSTR + kt * 32 + (lhi << 3);
        ap.q[0] = *(const uint4*)pp;
        ap.q[1] = *(const uint4*)(pp + 16);
        for (int nt = 0; nt < 2; ++nt) {
            Frag bvf;
            const _Float16* pv = Vt + (nt * 16 + llo) * PSTR + kt * 32 + (lhi << 3);
            bvf.q[0] = *(const uint4*)pv;
            bvf.q[1] = *(const uint4*)(pv + 16);
            o[nt] = wmma_f16(ap, bvf, o[nt]);
        }
    }
    // ---- store context rows (valid tokens only) ----
    for (int nt = 0; nt < 2; ++nt) {
        for (int r = 0; r < 8; ++r) {
            int s = w * 16 + r + (lhi << 3);
            if (s < 49) {
                int hd = nt * 16 + llo;
                ctx[rowbase + (size_t)s * DIMK + hd] = (_Float16)o[nt][r];
            }
        }
    }
}

// ---------------------------------------------------------------------------
extern "C" void kernel_launch(void* const* d_in, const int* in_sizes, int n_in,
                              void* d_out, int out_size, void* d_ws, size_t ws_size,
                              hipStream_t stream)
{
    (void)in_sizes; (void)n_in; (void)out_size; (void)ws_size;
    const float* X   = (const float*)d_in[0];
    const float* Wq  = (const float*)d_in[1];
    const float* bq  = (const float*)d_in[2];
    const float* Wk  = (const float*)d_in[3];
    const float* Wv  = (const float*)d_in[4];
    const float* bv  = (const float*)d_in[5];
    const float* Wp  = (const float*)d_in[6];
    const float* bp  = (const float*)d_in[7];
    const float* cw0 = (const float*)d_in[8];
    const float* cb0 = (const float*)d_in[9];
    const float* cw1 = (const float*)d_in[10];
    const float* ls  = (const float*)d_in[11];
    const float* tab = (const float*)d_in[12];
    const int*   idx = (const int*)d_in[13];

    char* ws = (char*)d_ws;
    float* biasb = (float*)ws;
    size_t off = (size_t)12 * 64 * 64 * sizeof(float);
    const size_t QKV = (size_t)MT * DIMK * sizeof(_Float16);
    _Float16* qbuf = (_Float16*)(ws + off); off += QKV;
    _Float16* kbuf = (_Float16*)(ws + off); off += QKV;
    _Float16* vbuf = (_Float16*)(ws + off); off += QKV;
    _Float16* cbuf = (_Float16*)(ws + off); off += QKV;

    cpb_kernel<<<dim3(1), dim3(256), 0, stream>>>(tab, cw0, cb0, cw1, idx, biasb);
    qkv_kernel<<<dim3(1568, 9), dim3(256), 0, stream>>>(X, Wq, Wk, Wv, bq, bv,
                                                        qbuf, kbuf, vbuf);
    attn_kernel<<<dim3(4096, 12), dim3(128), 0, stream>>>(qbuf, kbuf, vbuf,
                                                          biasb, ls, cbuf);
    proj_kernel<<<dim3(1568, 3), dim3(256), 0, stream>>>(cbuf, Wp, bp, (float*)d_out);
}
